// _SelfAttention_47656957116590
// MI455X (gfx1250) — matrix operand order repaired and checked by reading, compiled, not run on hardware
//
#include <hip/hip_runtime.h>

// Self-attention (b=2, s=2048, dim=1024, 16 heads x 64) for gfx1250 (MI455X).
// - v_wmma_f32_16x16x32_f16 for all GEMMs (f16 operands, f32 accumulate)
// - register double-buffered GEMM mainloops (partial s_wait_loadcnt overlap)
// - block-cooperative flash attention: 8 waves share K/V tiles staged into
//   LDS with global_load_async_to_lds_b128 (ASYNCcnt) and double buffering.
// Workspace: ~48 MB of f16 intermediates.

typedef __attribute__((ext_vector_type(16))) _Float16 v16h;
typedef __attribute__((ext_vector_type(8)))  _Float16 v8h;
typedef __attribute__((ext_vector_type(4)))  _Float16 v4h;
typedef __attribute__((ext_vector_type(8)))  float    v8f;

#define DIM   1024
#define NH    16
#define HD    64
#define SEQ   2048
#define BATCH 2
#define MROWS (BATCH * SEQ)        /* 4096 */
#define QK_SCALE 0.125f            /* 64^-0.5 */

__device__ __forceinline__ v8f wmma32(v16h a, v16h b, v8f c) {
  // (neg_a, A, neg_b, B, c_mod, C, reuse_a, reuse_b)
  return __builtin_amdgcn_wmma_f32_16x16x32_f16(false, a, false, b, (short)0, c,
                                                false, false);
}

// A-fragment (16x32 f16, M x K): lane holds row M = lane%16, K chunks
// [8*(lane/16) .. +7] and [+16 .. +23]  (ISA 7.12.2 "16-bit A-Matrix 16x32").
__device__ __forceinline__ v16h load_a_frag(const _Float16* __restrict__ p,
                                            int ld, int lane) {
  const _Float16* q = p + (size_t)(lane & 15) * ld + ((lane >> 4) << 3);
  v8h lo = *reinterpret_cast<const v8h*>(q);
  v8h hi = *reinterpret_cast<const v8h*>(q + 16);
  v16h f;
#pragma unroll
  for (int i = 0; i < 8; ++i) { f[i] = lo[i]; f[i + 8] = hi[i]; }
  return f;
}

// B-fragment (32x16 f16, K x N) loaded from B^T stored [N, K] row-major:
// lane holds column N = lane%16, K chunks [8*(lane/16) .. +7] and [+16 .. +23].
__device__ __forceinline__ v16h load_b_frag(const _Float16* p, int ld, int lane) {
  const _Float16* q = p + (size_t)(lane & 15) * ld + ((lane >> 4) << 3);
  v8h lo = *reinterpret_cast<const v8h*>(q);
  v8h hi = *reinterpret_cast<const v8h*>(q + 16);
  v16h f;
#pragma unroll
  for (int i = 0; i < 8; ++i) { f[i] = lo[i]; f[i + 8] = hi[i]; }
  return f;
}

// Async global -> LDS copy of 16 bytes per lane (GLOBAL_LOAD_ASYNC_TO_LDS_B128,
// tracked by ASYNCcnt).  Generic LDS addr low 32 bits == LDS byte address.
__device__ __forceinline__ void async_ld_b128(const _Float16* gaddr, _Float16* laddr) {
  unsigned l = (unsigned)(uintptr_t)laddr;
  asm volatile("global_load_async_to_lds_b128 %0, %1, off"
               :: "v"(l), "v"(gaddr) : "memory");
}
__device__ __forceinline__ void wait_async0() {
  asm volatile("s_wait_asynccnt 0x0" ::: "memory");
}

// ---------------- conversion kernels ----------------
__global__ __launch_bounds__(256) void cvt_f32_to_f16_x4(const float4* __restrict__ in,
                                                         v4h* __restrict__ out,
                                                         int n4) {
  int i = blockIdx.x * 256 + threadIdx.x;
  if (i < n4) {
    float4 v = in[i];
    v4h h = {(_Float16)v.x, (_Float16)v.y, (_Float16)v.z, (_Float16)v.w};
    out[i] = h;
  }
}

// LDS-tiled transpose+convert: out[n*K + k] = (f16) in[k*N + n].
__global__ __launch_bounds__(256) void transpose_cvt_f16(const float* __restrict__ in,
                                                         _Float16* __restrict__ out,
                                                         int K, int N) {
  __shared__ float tile[32][33];
  const int tx = threadIdx.x & 31;
  const int ty = threadIdx.x >> 5;                // 0..7
  const int ktiles = K >> 5;
  const int k0 = (blockIdx.x % ktiles) << 5;
  const int n0 = (blockIdx.x / ktiles) << 5;
#pragma unroll
  for (int r = 0; r < 4; ++r)
    tile[ty * 4 + r][tx] = in[(size_t)(k0 + ty * 4 + r) * N + n0 + tx];
  __syncthreads();
#pragma unroll
  for (int r = 0; r < 4; ++r)
    out[(size_t)(n0 + ty * 4 + r) * K + k0 + tx] = (_Float16)tile[tx][ty * 4 + r];
}

// ---------------- QKV projection GEMM ----------------
// C[4096,3072] = Xh @ Wqkv + b.  One wave = 32x64 tile, register double-buffered.
// Epilogue scatters: Q,K -> [b,h,s,d] f16 ; V -> V^T [b,h,d,s] f16.
__global__ __launch_bounds__(256) void qkv_gemm(const _Float16* __restrict__ xh,
                                                const _Float16* __restrict__ wqkvT,
                                                const float* __restrict__ bias,
                                                _Float16* __restrict__ qb,
                                                _Float16* __restrict__ kb,
                                                _Float16* __restrict__ vtb) {
  const int lane = threadIdx.x & 31;
  const int wave = blockIdx.x * 8 + (threadIdx.x >> 5);
  const int mtile = wave & 127;        // 4096/32 = 128
  const int nquad = wave >> 7;         // 3072/64 = 48
  const int m0 = mtile << 5;
  const int n0 = nquad << 6;

  const _Float16* arow0 = xh + (size_t)m0 * DIM;
  const _Float16* arow1 = arow0 + (size_t)16 * DIM;
  const _Float16* brow = wqkvT + (size_t)n0 * DIM;

  v8f acc[2][4] = {};
  v16h a0 = load_a_frag(arow0, DIM, lane);
  v16h a1 = load_a_frag(arow1, DIM, lane);
  v16h b0 = load_b_frag(brow, DIM, lane);
  v16h b1 = load_b_frag(brow + (size_t)16 * DIM, DIM, lane);
  v16h b2 = load_b_frag(brow + (size_t)32 * DIM, DIM, lane);
  v16h b3 = load_b_frag(brow + (size_t)48 * DIM, DIM, lane);

  for (int k0 = 0; k0 < DIM; k0 += 32) {
    const int kn = (k0 + 32 < DIM) ? (k0 + 32) : k0;   // branch-free clamp
    v16h na0 = load_a_frag(arow0 + kn, DIM, lane);
    v16h na1 = load_a_frag(arow1 + kn, DIM, lane);
    v16h nb0 = load_b_frag(brow + kn, DIM, lane);
    v16h nb1 = load_b_frag(brow + (size_t)16 * DIM + kn, DIM, lane);
    v16h nb2 = load_b_frag(brow + (size_t)32 * DIM + kn, DIM, lane);
    v16h nb3 = load_b_frag(brow + (size_t)48 * DIM + kn, DIM, lane);

    acc[0][0] = wmma32(a0, b0, acc[0][0]);
    acc[1][0] = wmma32(a1, b0, acc[1][0]);
    acc[0][1] = wmma32(a0, b1, acc[0][1]);
    acc[1][1] = wmma32(a1, b1, acc[1][1]);
    acc[0][2] = wmma32(a0, b2, acc[0][2]);
    acc[1][2] = wmma32(a1, b2, acc[1][2]);
    acc[0][3] = wmma32(a0, b3, acc[0][3]);
    acc[1][3] = wmma32(a1, b3, acc[1][3]);

    a0 = na0; a1 = na1; b0 = nb0; b1 = nb1; b2 = nb2; b3 = nb3;
  }

  const int which = n0 >> 10;          // 0:Q 1:K 2:V (uniform per wave)
  const int h = (n0 & 1023) >> 6;      // head (uniform per wave)
#pragma unroll
  for (int t = 0; t < 4; ++t) {
    const int n = n0 + t * 16 + (lane & 15);
    const float bn = bias[n];
    const int d = n & 63;
#pragma unroll
    for (int i = 0; i < 2; ++i) {
#pragma unroll
      for (int v = 0; v < 8; ++v) {
        const int m = m0 + 16 * i + v + ((lane >> 4) << 3);
        const int bidx = m >> 11, s = m & 2047;
        const _Float16 hv = (_Float16)(acc[i][t][v] + bn);
        const size_t bh = (size_t)(bidx * NH + h);
        if (which == 0)      qb[(bh * SEQ + s) * HD + d] = hv;
        else if (which == 1) kb[(bh * SEQ + s) * HD + d] = hv;
        else                 vtb[(bh * HD + d) * SEQ + s] = hv;
      }
    }
  }
}

// ---------------- flash attention (block-cooperative) ----------------
// One block (8 waves) = 8 q-tiles (128 rows) of one (b,h).  Each 32-key
// K-tile (32x64) and V^T-tile (64x32) is async-staged into LDS once and
// shared by all 8 waves, double-buffered against the compute.
__global__ __launch_bounds__(256) void flash_attn(const _Float16* __restrict__ qb,
                                                  const _Float16* __restrict__ kb,
                                                  const _Float16* __restrict__ vtb,
                                                  _Float16* __restrict__ oh) {
  __shared__ __align__(16) _Float16 kst[2][32 * HD];   // 2 x 4 KB
  __shared__ __align__(16) _Float16 vst[2][HD * 32];   // 2 x 4 KB
  __shared__ __align__(16) _Float16 pbuf[8][16 * 32];  // per-wave P transpose

  const int lane = threadIdx.x & 31;
  const int wslot = threadIdx.x >> 5;
  const int bh_i = blockIdx.x >> 4;          // 0..31  (b*16 + h)
  const int qt = ((blockIdx.x & 15) << 3) + wslot;   // q-tile 0..127
  const int h = bh_i & 15;
  const int bidx = bh_i >> 4;
  const size_t bh = (size_t)bh_i;

  const _Float16* Q = qb + (bh * SEQ + (size_t)qt * 16) * HD;
  const _Float16* K = kb + bh * SEQ * HD;
  const _Float16* V = vtb + bh * HD * SEQ;   // V^T: [d, s]
  _Float16* pb = &pbuf[wslot][0];

  // per-thread staging coordinates (16 B per thread per tile)
  const int krow = threadIdx.x >> 3, kcol = (threadIdx.x & 7) << 3;   // 32 x 64
  const int vrow = threadIdx.x >> 2, vcol = (threadIdx.x & 3) << 3;   // 64 x 32

  const v16h qf0 = load_a_frag(Q, HD, lane);        // d 0..31
  const v16h qf1 = load_a_frag(Q + 32, HD, lane);   // d 32..63

  v8f o[4] = {};
  float mrow[8], lrow[8];
#pragma unroll
  for (int v = 0; v < 8; ++v) { mrow[v] = -__builtin_inff(); lrow[v] = 0.f; }

  // stage tile 0
  async_ld_b128(K + (size_t)krow * HD + kcol, &kst[0][krow * HD + kcol]);
  async_ld_b128(V + (size_t)vrow * SEQ + vcol, &vst[0][vrow * 32 + vcol]);
  wait_async0();
  __syncthreads();

  int cur = 0;
  for (int ks = 0; ks < SEQ; ks += 32) {
    // async-stage the next tile into the other buffer while we compute
    if (ks + 32 < SEQ) {
      async_ld_b128(K + (size_t)(ks + 32 + krow) * HD + kcol,
                    &kst[cur ^ 1][krow * HD + kcol]);
      async_ld_b128(V + (size_t)vrow * SEQ + (ks + 32) + vcol,
                    &vst[cur ^ 1][vrow * 32 + vcol]);
    }

    // S = Q @ K^T for 32 keys (two 16-col sub-tiles), K-fragments from LDS
    const _Float16* kt = &kst[cur][0];
    v8f s0 = {}, s1 = {};
    s0 = wmma32(qf0, load_b_frag(kt, HD, lane), s0);
    s0 = wmma32(qf1, load_b_frag(kt + 32, HD, lane), s0);
    s1 = wmma32(qf0, load_b_frag(kt + 16 * HD, HD, lane), s1);
    s1 = wmma32(qf1, load_b_frag(kt + 16 * HD + 32, HD, lane), s1);

    // online softmax; VGPR index v owns row (v + 8*(lane/16)); a row's 16
    // columns live across the 16 lanes of that half-wave.
#pragma unroll
    for (int v = 0; v < 8; ++v) {
      float a0 = s0[v] * QK_SCALE, a1 = s1[v] * QK_SCALE;
      float t = fmaxf(a0, a1);
      t = fmaxf(t, __shfl_xor(t, 1, 32));
      t = fmaxf(t, __shfl_xor(t, 2, 32));
      t = fmaxf(t, __shfl_xor(t, 4, 32));
      t = fmaxf(t, __shfl_xor(t, 8, 32));
      const float mn = fmaxf(mrow[v], t);
      const float alpha = __expf(mrow[v] - mn);
      mrow[v] = mn;
      const float p0 = __expf(a0 - mn), p1 = __expf(a1 - mn);
      float rs = p0 + p1;
      rs += __shfl_xor(rs, 1, 32);
      rs += __shfl_xor(rs, 2, 32);
      rs += __shfl_xor(rs, 4, 32);
      rs += __shfl_xor(rs, 8, 32);
      lrow[v] = lrow[v] * alpha + rs;
#pragma unroll
      for (int t4 = 0; t4 < 4; ++t4) o[t4][v] *= alpha;
      // stage P (D-layout) into per-wave LDS for transpose to A-layout
      const int row = v + ((lane >> 4) << 3);
      pb[row * 32 + (lane & 15)]      = (_Float16)p0;
      pb[row * 32 + 16 + (lane & 15)] = (_Float16)p1;
    }
    asm volatile("" ::: "memory");          // keep LDS store->load order (hw in-order)
    const v16h pf = load_a_frag(pb, 32, lane);
    asm volatile("" ::: "memory");

    // O += P @ V  (B-fragments from LDS V^T tile, contiguous along keys)
    const _Float16* vt = &vst[cur][0];
#pragma unroll
    for (int t4 = 0; t4 < 4; ++t4)
      o[t4] = wmma32(pf, load_b_frag(vt + (size_t)(t4 * 16) * 32, 32, lane), o[t4]);

    // hand off: my async stage done, everyone done reading 'cur'
    wait_async0();
    __syncthreads();
    cur ^= 1;
  }

  // normalize and write [b, s, h*64+d] f16 (row-major [4096, 1024] A for proj)
#pragma unroll
  for (int t4 = 0; t4 < 4; ++t4) {
#pragma unroll
    for (int v = 0; v < 8; ++v) {
      const int s = qt * 16 + v + ((lane >> 4) << 3);
      const int col = h * HD + t4 * 16 + (lane & 15);
      oh[((size_t)bidx * SEQ + s) * DIM + col] = (_Float16)(o[t4][v] / lrow[v]);
    }
  }
}

// ---------------- output projection GEMM ----------------
// out[4096,1024] = Ah @ Wproj + b ; one wave = 32x64 tile, double-buffered.
__global__ __launch_bounds__(256) void proj_gemm(const _Float16* __restrict__ ah,
                                                 const _Float16* __restrict__ wprojT,
                                                 const float* __restrict__ bias,
                                                 float* __restrict__ out) {
  const int lane = threadIdx.x & 31;
  const int wave = blockIdx.x * 8 + (threadIdx.x >> 5);
  const int mtile = wave & 127;        // 128 m-tiles of 32
  const int nquad = wave >> 7;         // 1024/64 = 16
  const int m0 = mtile << 5;
  const int n0 = nquad << 6;

  const _Float16* arow0 = ah + (size_t)m0 * DIM;
  const _Float16* arow1 = arow0 + (size_t)16 * DIM;
  const _Float16* brow = wprojT + (size_t)n0 * DIM;

  v8f acc[2][4] = {};
  v16h a0 = load_a_frag(arow0, DIM, lane);
  v16h a1 = load_a_frag(arow1, DIM, lane);
  v16h b0 = load_b_frag(brow, DIM, lane);
  v16h b1 = load_b_frag(brow + (size_t)16 * DIM, DIM, lane);
  v16h b2 = load_b_frag(brow + (size_t)32 * DIM, DIM, lane);
  v16h b3 = load_b_frag(brow + (size_t)48 * DIM, DIM, lane);

  for (int k0 = 0; k0 < DIM; k0 += 32) {
    const int kn = (k0 + 32 < DIM) ? (k0 + 32) : k0;
    v16h na0 = load_a_frag(arow0 + kn, DIM, lane);
    v16h na1 = load_a_frag(arow1 + kn, DIM, lane);
    v16h nb0 = load_b_frag(brow + kn, DIM, lane);
    v16h nb1 = load_b_frag(brow + (size_t)16 * DIM + kn, DIM, lane);
    v16h nb2 = load_b_frag(brow + (size_t)32 * DIM + kn, DIM, lane);
    v16h nb3 = load_b_frag(brow + (size_t)48 * DIM + kn, DIM, lane);

    acc[0][0] = wmma32(a0, b0, acc[0][0]);
    acc[1][0] = wmma32(a1, b0, acc[1][0]);
    acc[0][1] = wmma32(a0, b1, acc[0][1]);
    acc[1][1] = wmma32(a1, b1, acc[1][1]);
    acc[0][2] = wmma32(a0, b2, acc[0][2]);
    acc[1][2] = wmma32(a1, b2, acc[1][2]);
    acc[0][3] = wmma32(a0, b3, acc[0][3]);
    acc[1][3] = wmma32(a1, b3, acc[1][3]);

    a0 = na0; a1 = na1; b0 = nb0; b1 = nb1; b2 = nb2; b3 = nb3;
  }

#pragma unroll
  for (int t = 0; t < 4; ++t) {
    const int n = n0 + t * 16 + (lane & 15);
    const float bn = bias[n];
#pragma unroll
    for (int i = 0; i < 2; ++i) {
#pragma unroll
      for (int v = 0; v < 8; ++v) {
        const int m = m0 + 16 * i + v + ((lane >> 4) << 3);
        out[(size_t)m * DIM + n] = acc[i][t][v] + bn;
      }
    }
  }
}

extern "C" void kernel_launch(void* const* d_in, const int* in_sizes, int n_in,
                              void* d_out, int out_size, void* d_ws, size_t ws_size,
                              hipStream_t stream) {
  const float* x      = (const float*)d_in[0];
  const float* w_qkv  = (const float*)d_in[1];
  const float* b_qkv  = (const float*)d_in[2];
  const float* w_proj = (const float*)d_in[3];
  const float* b_proj = (const float*)d_in[4];
  float* out = (float*)d_out;

  // workspace carve-up (f16 elements); total = 25,165,824 elems = 48 MiB
  _Float16* p = (_Float16*)d_ws;
  _Float16* xh     = p; p += (size_t)MROWS * DIM;       // 4096x1024
  _Float16* wqkvT  = p; p += (size_t)3 * DIM * DIM;     // 3072x1024 (W^T)
  _Float16* wprojT = p; p += (size_t)DIM * DIM;         // 1024x1024 (W^T)
  _Float16* qb     = p; p += (size_t)MROWS * DIM;       // [b,h,s,d]
  _Float16* kb     = p; p += (size_t)MROWS * DIM;       // [b,h,s,d]
  _Float16* vtb    = p; p += (size_t)MROWS * DIM;       // [b,h,d,s]
  _Float16* ah     = p;                                 // 4096x1024

  const int nx4 = (MROWS * DIM) / 4;
  cvt_f32_to_f16_x4<<<(nx4 + 255) / 256, 256, 0, stream>>>((const float4*)x, (v4h*)xh, nx4);
  transpose_cvt_f16<<<(DIM / 32) * (3 * DIM / 32), 256, 0, stream>>>(w_qkv, wqkvT, DIM, 3 * DIM);
  transpose_cvt_f16<<<(DIM / 32) * (DIM / 32), 256, 0, stream>>>(w_proj, wprojT, DIM, DIM);

  qkv_gemm<<<(128 * 48) / 8, 256, 0, stream>>>(xh, wqkvT, b_qkv, qb, kb, vtb);
  flash_attn<<<(BATCH * NH * (SEQ / 16)) / 8, 256, 0, stream>>>(qb, kb, vtb, ah);
  proj_gemm<<<(128 * 16) / 8, 256, 0, stream>>>(ah, wprojT, b_proj, out);
}
